// SampleCrossAttention_31980326486249
// MI455X (gfx1250) — matrix operand order repaired and checked by reading, not compiled, not run on hardware
//
#include <hip/hip_runtime.h>
#include <hip/hip_bf16.h>

#define HW      16384
#define BATCH   4
#define DIM     256
#define HEADS   8
#define HD      32
#define NSLOT   256
#define POSK    512
#define NSK     768
#define LN_EPS  1e-5f

typedef __attribute__((ext_vector_type(16))) _Float16 v16h;
typedef __attribute__((ext_vector_type(8)))  float    v8f;

// ---------------------------------------------------------------------------
// Kernel 0: convert Wk, Wv (rows [D,3D) of in_proj_weight) to f16
// ---------------------------------------------------------------------------
__global__ __launch_bounds__(256) void cvt_w_kernel(
    const float* __restrict__ Wfull, _Float16* __restrict__ Wk_h,
    _Float16* __restrict__ Wv_h) {
  int i = blockIdx.x * 256 + threadIdx.x;              // 0..65535
  Wk_h[i] = (_Float16)Wfull[DIM * DIM + i];
  Wv_h[i] = (_Float16)Wfull[2 * DIM * DIM + i];
}

// ---------------------------------------------------------------------------
// Kernel 1: per-slot top-768 descending argsort (radix select + bitonic)
// ---------------------------------------------------------------------------
__device__ inline unsigned tokey(float f) {
  unsigned u = __float_as_uint(f);
  return u ^ ((u & 0x80000000u) ? 0xFFFFFFFFu : 0x80000000u);
}

__global__ __launch_bounds__(256) void topk_kernel(
    const float* __restrict__ curio, int* __restrict__ idx_ws) {
  int s = blockIdx.x, t = threadIdx.x;
  const float* vals = curio + (size_t)s * HW;
  __shared__ unsigned hist[256];
  __shared__ unsigned sh_prefix, sh_want, sh_cnt_gt, sh_cnt_eq;
  __shared__ unsigned long long arr[1024];

  if (t == 0) { sh_prefix = 0; sh_want = NSK; }
  __syncthreads();

  // 4-pass radix select (MSB first) for the rank-768 threshold key
  for (int shift = 24; shift >= 0; shift -= 8) {
    hist[t] = 0;
    __syncthreads();
    unsigned prefix = sh_prefix;
    for (int i = t; i < HW; i += 256) {
      unsigned key = tokey(vals[i]);
      if (shift == 24 || (key >> (shift + 8)) == prefix)
        atomicAdd(&hist[(key >> shift) & 255u], 1u);
    }
    __syncthreads();
    if (t == 0) {
      unsigned want = sh_want, acc = 0;
      int g = 0;
      for (int d = 255; d >= 0; --d) {
        if (acc + hist[d] >= want) { g = d; break; }
        acc += hist[d];
      }
      sh_prefix = (prefix << 8) | (unsigned)g;
      sh_want = want - acc;            // how many ==T entries still needed
    }
    __syncthreads();
  }

  if (t == 0) { sh_cnt_gt = 0; sh_cnt_eq = (unsigned)NSK - sh_want; }
  __syncthreads();
  unsigned T = sh_prefix;

  // collect: keys > T first, then enough keys == T
  for (int i = t; i < HW; i += 256) {
    unsigned key = tokey(vals[i]);
    if (key > T) {
      unsigned p = atomicAdd(&sh_cnt_gt, 1u);
      arr[p] = ((unsigned long long)(~key) << 32) | (unsigned)i;
    } else if (key == T) {
      unsigned p = atomicAdd(&sh_cnt_eq, 1u);
      if (p < NSK)
        arr[p] = ((unsigned long long)(~key) << 32) | (unsigned)i;
    }
  }
  __syncthreads();
  for (int i = NSK + t; i < 1024; i += 256) arr[i] = ~0ULL;
  __syncthreads();

  // bitonic ascending sort of (~key, idx): descending value, stable by index
  for (int k = 2; k <= 1024; k <<= 1) {
    for (int j = k >> 1; j > 0; j >>= 1) {
      for (int i = t; i < 1024; i += 256) {
        int ixj = i ^ j;
        if (ixj > i) {
          unsigned long long a = arr[i], b = arr[ixj];
          bool up = (i & k) == 0;
          if ((a > b) == up) { arr[i] = b; arr[ixj] = a; }
        }
      }
      __syncthreads();
    }
  }
  for (int i = t; i < NSK; i += 256)
    idx_ws[s * NSK + i] = (int)(arr[i] & 0xFFFFFFFFu);
}

// ---------------------------------------------------------------------------
// Kernel 2: q = (slots @ Wq^T + bq) * HD^-0.5   (tiny; VALU)
// ---------------------------------------------------------------------------
__global__ __launch_bounds__(256) void q_kernel(
    const float* __restrict__ slots, const float* __restrict__ Wfull,
    const float* __restrict__ bias, float* __restrict__ q_ws) {
  int s = blockIdx.x, e = threadIdx.x;
  __shared__ float srow[DIM];
  srow[e] = slots[s * DIM + e];
  __syncthreads();
  const float* w = Wfull + (size_t)e * DIM;  // Wq rows = first D rows
  float acc = 0.f;
  for (int d = 0; d < DIM; ++d) acc += srow[d] * w[d];
  q_ws[s * DIM + e] = (acc + bias[e]) * 0.17677669529663687f; // 1/sqrt(32)
}

// ---------------------------------------------------------------------------
// Kernel 3: gather rows, K/V projection via WMMA f16, fused QK -> aff
//   block = (m-tile of 16 keys, slot); 256 thr = 8 waves; wave w == head w
// ---------------------------------------------------------------------------
__global__ __launch_bounds__(256) void proj_kernel(
    const float* __restrict__ features, const float* __restrict__ pos_enc,
    const int* __restrict__ batch_idx, const int* __restrict__ idx_ws,
    const _Float16* __restrict__ Wk_h, const _Float16* __restrict__ Wv_h,
    const float* __restrict__ in_bias, const float* __restrict__ q_ws,
    float* __restrict__ aff_ws, _Float16* __restrict__ v_ws) {
  const int tile = blockIdx.x;      // 0..47
  const int s    = blockIdx.y;      // slot
  const int m0   = tile * 16;
  const bool doV = (m0 < POSK);

  __shared__ _Float16 Akeys[16][DIM + 8];
  __shared__ _Float16 Avals[16][DIM + 8];

  const int t = threadIdx.x;
  const int b = batch_idx[s];
  {
    int r = t >> 4;               // row 0..15
    int c = (t & 15) * 16;        // 16-dim chunk
    int gi = idx_ws[s * NSK + m0 + r];
    const float* frow = features + ((size_t)gi * BATCH + b) * DIM;
    const float* prow = pos_enc  + ((size_t)gi * BATCH + b) * DIM;
#pragma unroll
    for (int j = 0; j < 16; ++j) {
      float fv = frow[c + j], pv = prow[c + j];
      Akeys[r][c + j] = (_Float16)(fv + pv);
      Avals[r][c + j] = (_Float16)fv;
    }
  }
  __syncthreads();

  const int wave = t >> 5;            // == head
  const int lane = t & 31;
  const int lrow = lane & 15;
  const bool hi  = lane >= 16;
  const int n0 = wave * 32;           // N-tile bases (within head `wave`)
  const int n1 = n0 + 16;

  v8f ck0 = {}, ck1 = {}, cv0 = {}, cv1 = {};

  for (int k0 = 0; k0 < DIM; k0 += 32) {
    // A fragment: lane = row, 16 contiguous K halves (matches B)
    v16h a_k, a_v;
    const int kb0 = k0 + (hi ? 16 : 0);
    const int kb1 = kb0 + 8;
#pragma unroll
    for (int j = 0; j < 8; ++j) {
      a_k[j]     = Akeys[lrow][kb0 + j];
      a_k[8 + j] = Akeys[lrow][kb1 + j];
      a_v[j]     = Avals[lrow][kb0 + j];
      a_v[8 + j] = Avals[lrow][kb1 + j];
    }
    // B fragment: lane = output column, 16 contiguous K halves
    const int koff = k0 + (hi ? 16 : 0);
    const _Float16* wk0 = Wk_h + (size_t)(n0 + lrow) * DIM + koff;
    const _Float16* wk1 = Wk_h + (size_t)(n1 + lrow) * DIM + koff;
    const _Float16* wv0 = Wv_h + (size_t)(n0 + lrow) * DIM + koff;
    const _Float16* wv1 = Wv_h + (size_t)(n1 + lrow) * DIM + koff;
    v16h b0k, b1k, b0v, b1v;
#pragma unroll
    for (int j = 0; j < 16; ++j) {
      b0k[j] = wk0[j]; b1k[j] = wk1[j];
      b0v[j] = wv0[j]; b1v[j] = wv1[j];
    }
    ck0 = __builtin_amdgcn_wmma_f32_16x16x32_f16(false, a_k, false, b0k,
                                                 (short)0, ck0, false, false);
    ck1 = __builtin_amdgcn_wmma_f32_16x16x32_f16(false, a_k, false, b1k,
                                                 (short)0, ck1, false, false);
    if (doV) {
      cv0 = __builtin_amdgcn_wmma_f32_16x16x32_f16(false, a_v, false, b0v,
                                                   (short)0, cv0, false, false);
      cv1 = __builtin_amdgcn_wmma_f32_16x16x32_f16(false, a_v, false, b1v,
                                                   (short)0, cv1, false, false);
    }
  }

  // fused QK: aff[s][m][head] = sum_e q[s][e] * (k[m][e] + bk[e])
  const float qv0 = q_ws[s * DIM + n0 + lrow];
  const float qv1 = q_ws[s * DIM + n1 + lrow];
  const float bk0 = in_bias[DIM + n0 + lrow];
  const float bk1 = in_bias[DIM + n1 + lrow];
  float part[8];
#pragma unroll
  for (int r = 0; r < 8; ++r)
    part[r] = qv0 * (ck0[r] + bk0) + qv1 * (ck1[r] + bk1);
#pragma unroll
  for (int m = 8; m >= 1; m >>= 1)
#pragma unroll
    for (int r = 0; r < 8; ++r) part[r] += __shfl_xor(part[r], m, 32);
  if (lane == 0 || lane == 16) {
    int mb = hi ? 8 : 0;
#pragma unroll
    for (int r = 0; r < 8; ++r)
      aff_ws[((size_t)s * NSK + m0 + mb + r) * HEADS + wave] = part[r];
  }

  if (doV) {
    const float bv0 = in_bias[2 * DIM + n0 + lrow];
    const float bv1 = in_bias[2 * DIM + n1 + lrow];
    int mb = hi ? 8 : 0;
#pragma unroll
    for (int r = 0; r < 8; ++r) {
      size_t base = ((size_t)s * POSK + m0 + mb + r) * DIM;
      v_ws[base + n0 + lrow] = (_Float16)(cv0[r] + bv0);
      v_ws[base + n1 + lrow] = (_Float16)(cv1[r] + bv1);
    }
  }
}

// ---------------------------------------------------------------------------
// Kernel 4: per-slot softmax, attn@V, out-proj, residual+LN, norm_aff
// ---------------------------------------------------------------------------
__global__ __launch_bounds__(256) void attn_kernel(
    const float* __restrict__ aff_ws, const _Float16* __restrict__ v_ws,
    const float* __restrict__ slots, const float* __restrict__ Wout,
    const float* __restrict__ bout, const float* __restrict__ ln_w,
    const float* __restrict__ ln_b, float* __restrict__ out) {
  int s = blockIdx.x, t = threadIdx.x;
  int wave = t >> 5, lane = t & 31;
  __shared__ float attnL[HEADS][POSK];
  __shared__ float outv[DIM];
  __shared__ float red[32];
  const float* affS = aff_ws + (size_t)s * NSK * HEADS;

  // softmax over first 512 keys; wave w == head w
  float loc[16];
  float mx = -1e30f;
#pragma unroll
  for (int i = 0; i < 16; ++i) {
    float a = affS[(lane + i * 32) * HEADS + wave];
    loc[i] = a;
    mx = fmaxf(mx, a);
  }
  for (int m = 16; m >= 1; m >>= 1) mx = fmaxf(mx, __shfl_xor(mx, m, 32));
  float ssum = 0.f;
#pragma unroll
  for (int i = 0; i < 16; ++i) { loc[i] = __expf(loc[i] - mx); ssum += loc[i]; }
  for (int m = 16; m >= 1; m >>= 1) ssum += __shfl_xor(ssum, m, 32);
  float inv = 1.0f / ssum;
#pragma unroll
  for (int i = 0; i < 16; ++i) attnL[wave][lane + i * 32] = loc[i] * inv;
  __syncthreads();

  // out[t] = sum_n attn[h][n] * v[s][n][t]
  {
    int h = t >> 5;
    const _Float16* vS = v_ws + (size_t)s * POSK * DIM + t;
    float acc = 0.f;
    for (int n = 0; n < POSK; ++n)
      acc += attnL[h][n] * (float)vS[(size_t)n * DIM];
    outv[t] = acc;
  }
  __syncthreads();

  // output projection + residual
  float o2 = bout[t];
  {
    const float* w = Wout + (size_t)t * DIM;
    for (int d = 0; d < DIM; ++d) o2 += outv[d] * w[d];
  }
  float x = slots[s * DIM + t] + o2;

  // LayerNorm over 256
  float sum = x, sq = x * x;
  for (int m = 16; m >= 1; m >>= 1) {
    sum += __shfl_xor(sum, m, 32);
    sq  += __shfl_xor(sq, m, 32);
  }
  if (lane == 0) { red[wave] = sum; red[8 + wave] = sq; }
  __syncthreads();
  float ts = 0.f, tq = 0.f;
  for (int wv = 0; wv < 8; ++wv) { ts += red[wv]; tq += red[8 + wv]; }
  float mean = ts * (1.0f / DIM);
  float var  = tq * (1.0f / DIM) - mean * mean;
  float rstd = rsqrtf(var + LN_EPS);
  out[s * DIM + t] = (x - mean) * rstd * ln_w[t] + ln_b[t];

  // norm_aff: per slot, LN over head-means of all 768 affinities
  __syncthreads();
  float am[3], s1 = 0.f, s2 = 0.f;
#pragma unroll
  for (int i = 0; i < 3; ++i) {
    int n = t + i * 256;
    float a = 0.f;
    for (int h = 0; h < HEADS; ++h) a += affS[n * HEADS + h];
    a *= 0.125f;
    am[i] = a; s1 += a; s2 += a * a;
  }
  for (int m = 16; m >= 1; m >>= 1) {
    s1 += __shfl_xor(s1, m, 32);
    s2 += __shfl_xor(s2, m, 32);
  }
  if (lane == 0) { red[16 + wave] = s1; red[24 + wave] = s2; }
  __syncthreads();
  float T1 = 0.f, T2 = 0.f;
  for (int wv = 0; wv < 8; ++wv) { T1 += red[16 + wv]; T2 += red[24 + wv]; }
  float amean = T1 * (1.0f / NSK);
  float arstd = rsqrtf(T2 * (1.0f / NSK) - amean * amean + LN_EPS);
#pragma unroll
  for (int i = 0; i < 3; ++i) {
    int n = t + i * 256;
    out[NSLOT * DIM + n * NSLOT + s] = (am[i] - amean) * arstd;
  }
}

// ---------------------------------------------------------------------------
extern "C" void kernel_launch(void* const* d_in, const int* in_sizes, int n_in,
                              void* d_out, int out_size, void* d_ws,
                              size_t ws_size, hipStream_t stream) {
  (void)in_sizes; (void)n_in; (void)out_size; (void)ws_size;
  const float* slots    = (const float*)d_in[0];
  const float* features = (const float*)d_in[1];
  const float* pos_enc  = (const float*)d_in[2];
  const float* curio    = (const float*)d_in[3];
  const int*   batch_idx= (const int*)d_in[4];
  const float* in_w     = (const float*)d_in[5];
  const float* in_b     = (const float*)d_in[6];
  const float* out_w    = (const float*)d_in[7];
  const float* out_b    = (const float*)d_in[8];
  const float* lnw      = (const float*)d_in[9];
  const float* lnb      = (const float*)d_in[10];
  float* out = (float*)d_out;

  char* wsb = (char*)d_ws;
  int*      idx_ws = (int*)wsb;                          //   786432 B
  float*    q_ws   = (float*)(wsb + 786432);             //   262144 B
  _Float16* Wk_h   = (_Float16*)(wsb + 1048576);         //   131072 B
  _Float16* Wv_h   = (_Float16*)(wsb + 1179648);         //   131072 B
  float*    aff_ws = (float*)(wsb + 1310720);            //  6291456 B
  _Float16* v_ws   = (_Float16*)(wsb + 7602176);         // 67108864 B

  cvt_w_kernel<<<256, 256, 0, stream>>>(in_w, Wk_h, Wv_h);
  topk_kernel<<<NSLOT, 256, 0, stream>>>(curio, idx_ws);
  q_kernel<<<NSLOT, 256, 0, stream>>>(slots, in_w, in_b, q_ws);
  proj_kernel<<<dim3(NSK / 16, NSLOT), 256, 0, stream>>>(
      features, pos_enc, batch_idx, idx_ws, Wk_h, Wv_h, in_b, q_ws, aff_ws,
      v_ws);
  attn_kernel<<<NSLOT, 256, 0, stream>>>(aff_ws, v_ws, slots, out_w, out_b,
                                         lnw, lnb, out);
}
